// SwitchingDynamicsParameter_5944234737894
// MI455X (gfx1250) — matrix operand order repaired and checked by reading, not compiled, not run on hardware
//
#include <hip/hip_runtime.h>
#include <hip/hip_bf16.h>

typedef __attribute__((ext_vector_type(16))) _Float16 v16h;
typedef __attribute__((ext_vector_type(8)))  float    v8f;
typedef __attribute__((ext_vector_type(4)))  float    v4f;

#define WMMA_F16(a, b, c) \
  __builtin_amdgcn_wmma_f32_16x16x32_f16(false, (a), false, (b), (short)0, (c), false, false)

// Problem constants (from reference)
constexpr int Pc  = 32;    // input dim
constexpr int Hc  = 32;    // GRU hidden
constexpr int Bsz = 128;   // batch
constexpr int Tn  = 512;   // sequence length

__device__ __forceinline__ float sigmoidf_(float x) { return 1.f / (1.f + __expf(-x)); }
__device__ __forceinline__ float tanhf_(float x) {
  x = fminf(fmaxf(x, -15.f), 15.f);
  float e = __expf(2.f * x);
  return (e - 1.f) / (e + 1.f);
}
__device__ __forceinline__ float gumbelf_(float u) { return -__logf(-__logf(u)); }

// ---------------------------------------------------------------------------
// K1: gx = a_seq @ wih^T + bih, both directions, stored in WMMA-C layout:
//   gx[(rt*6 + nt)*256 + lane*8 + v]   with rt = t*8 + mtile
// so the GRU kernel can re-load it with coalesced b128 loads in the exact
// per-lane pattern its accumulators use. One wave = one 16x16 tile (K=32).
// grid = 4096 row-tiles * 6 col-tiles * 2 dirs, block = 32 (one wave)
// ---------------------------------------------------------------------------
__global__ void gx_kernel(const float* __restrict__ a_seq,
                          const float* __restrict__ wih_f, const float* __restrict__ bih_f,
                          const float* __restrict__ wih_b, const float* __restrict__ bih_b,
                          float* __restrict__ gxf, float* __restrict__ gxb) {
  int bid = blockIdx.x;
  int rt  = bid / 12;
  int rem = bid % 12;
  int nt  = rem % 6;
  int dir = rem / 6;
  int l = threadIdx.x, nl = l & 15, half = l >> 4;

  // A fragment: rows r = rt*16 + (lane&15), K contiguous per lane half (k = 16*half + s)
  int r = rt * 16 + nl;
  int t = r >> 7, b = r & 127;
  const float* arow = a_seq + ((size_t)b * Tn + t) * Pc;
  v16h a;
#pragma unroll
  for (int v = 0; v < 8; ++v) {
    int kb = 2 * v + 16 * half;
    a[2 * v]     = (_Float16)arow[kb];
    a[2 * v + 1] = (_Float16)arow[kb + 1];
  }

  // B fragment: B[k][n] = wih[n][k]
  const float* wih = dir ? wih_b : wih_f;
  const float* bih = dir ? bih_b : bih_f;
  int n = nt * 16 + nl;
  v16h bf;
#pragma unroll
  for (int v = 0; v < 8; ++v) {
    int kb = 2 * v + 16 * half;
    bf[2 * v]     = (_Float16)wih[n * Pc + kb];
    bf[2 * v + 1] = (_Float16)wih[n * Pc + kb + 1];
  }

  v8f c = {};
  c = WMMA_F16(a, bf, c);

  float bias = bih[n];
  float* gx = dir ? gxb : gxf;
  float* dst = gx + ((size_t)rt * 6 + nt) * 256 + l * 8;
  v4f lo = {c[0] + bias, c[1] + bias, c[2] + bias, c[3] + bias};
  v4f hi = {c[4] + bias, c[5] + bias, c[6] + bias, c[7] + bias};
  *(v4f*)(dst)     = lo;
  *(v4f*)(dst + 4) = hi;
}

// ---------------------------------------------------------------------------
// K2: bidirectional GRU recurrence. 16 fully independent single-wave blocks:
// dir (2) x batch-tile of 16 rows (8). whh B-fragments live in VGPRs for the
// whole 512-step scan; h round-trips through a 1KB LDS transpose buffer.
// gx arrives pre-permuted in C-layout -> 12 b128 loads per step, issued
// independently of the h-dependent WMMA chain.
// ---------------------------------------------------------------------------
__global__ void gru_kernel(const float* __restrict__ whh_f, const float* __restrict__ bhh_f,
                           const float* __restrict__ whh_b, const float* __restrict__ bhh_b,
                           const float* __restrict__ gxf, const float* __restrict__ gxb,
                           _Float16* __restrict__ hseq) {
  int dir = blockIdx.x >> 3, mt = blockIdx.x & 7;
  const float* whh = dir ? whh_b : whh_f;
  const float* bhh = dir ? bhh_b : bhh_f;
  const float* gx  = dir ? gxb : gxf;
  int l = threadIdx.x, nl = l & 15, half = l >> 4;

  // Resident B-fragments for the 6 N-tiles of whh^T (N = 96 = 3 gates * 32)
  v16h Bw[6];
#pragma unroll
  for (int g = 0; g < 3; ++g)
#pragma unroll
    for (int jh = 0; jh < 2; ++jh) {
      int nt = 2 * g + jh;
      int n  = nt * 16 + nl;
#pragma unroll
      for (int v = 0; v < 8; ++v) {
        int kb = 2 * v + 16 * half;
        Bw[nt][2 * v]     = (_Float16)whh[n * Hc + kb];
        Bw[nt][2 * v + 1] = (_Float16)whh[n * Hc + kb + 1];
      }
    }
  // Per-lane hidden-to-hidden biases (column is fixed per lane in C layout)
  float bR[2], bZ[2], bN[2];
#pragma unroll
  for (int jh = 0; jh < 2; ++jh) {
    int j = nl + 16 * jh;
    bR[jh] = bhh[j]; bZ[jh] = bhh[32 + j]; bN[jh] = bhh[64 + j];
  }

  float h[8][2];
#pragma unroll
  for (int v = 0; v < 8; ++v) { h[v][0] = 0.f; h[v][1] = 0.f; }

  __shared__ _Float16 sh[16 * 32];  // h tile, [local row][hidden], f16
  for (int i = l; i < 512; i += 32) sh[i] = (_Float16)0.f;
  __syncthreads();

  for (int step = 0; step < Tn; ++step) {
    int ts = dir ? (Tn - 1 - step) : step;

    // gx tiles for this step: C-layout, contiguous per lane (12 b128 loads,
    // independent of h -> overlaps the WMMA chain)
    const float* gxr = gx + ((size_t)(ts * 8 + mt) * 6) * 256 + l * 8;
    v4f xg[6][2];
#pragma unroll
    for (int i = 0; i < 6; ++i) {
      xg[i][0] = *(const v4f*)(gxr + i * 256);
      xg[i][1] = *(const v4f*)(gxr + i * 256 + 4);
    }

    // A fragment of h from LDS (transpose C-layout rows -> A-layout rows)
    v16h a;
#pragma unroll
    for (int v = 0; v < 8; ++v) {
      int kb = 2 * v + 16 * half;
      a[2 * v]     = sh[nl * 32 + kb];
      a[2 * v + 1] = sh[nl * 32 + kb + 1];
    }

    v8f acc[6];
#pragma unroll
    for (int i = 0; i < 6; ++i) {
      v8f z = {};
      acc[i] = WMMA_F16(a, Bw[i], z);
    }
    __syncthreads();  // all sh reads done before overwrite

    // Gate math: r/z/n for element (row,j) share (lane, v) across acc tiles,
    // and the matching gx values share the same (lane, v) in xg.
#pragma unroll
    for (int v = 0; v < 8; ++v) {
      int mc  = v + 8 * half;       // local row in tile
      int row = mt * 16 + mc;       // global batch row
#pragma unroll
      for (int jh = 0; jh < 2; ++jh) {
        int j = nl + 16 * jh;
        float xr = xg[0 + jh][v >> 2][v & 3];
        float xz = xg[2 + jh][v >> 2][v & 3];
        float xn = xg[4 + jh][v >> 2][v & 3];
        float hr = acc[0 + jh][v] + bR[jh];
        float hz = acc[2 + jh][v] + bZ[jh];
        float hn = acc[4 + jh][v] + bN[jh];
        float rr = sigmoidf_(xr + hr);
        float zz = sigmoidf_(xz + hz);
        float nn = tanhf_(xn + rr * hn);
        float hv = (1.f - zz) * nn + zz * h[v][jh];
        h[v][jh] = hv;
        sh[mc * 32 + j] = (_Float16)hv;
        hseq[((size_t)ts * Bsz + row) * 64 + dir * 32 + j] = (_Float16)hv;
      }
    }
    __syncthreads();
  }
}

// ---------------------------------------------------------------------------
// K3: logits[t][b][64] = h_seq @ wl^T + bl.  K=64 -> two accumulated WMMAs.
// grid = 4096 row-tiles * 4 col-tiles, block = 32 (one wave)
// ---------------------------------------------------------------------------
__global__ void logits_kernel(const _Float16* __restrict__ hseq,
                              const float* __restrict__ wl, const float* __restrict__ bl,
                              float* __restrict__ logits) {
  int rt = blockIdx.x >> 2, nt = blockIdx.x & 3;
  int l = threadIdx.x, nl = l & 15, half = l >> 4;
  int r = rt * 16 + nl;
  const _Float16* hrow = hseq + (size_t)r * 64;
  int n = nt * 16 + nl;

  v8f c = {};
#pragma unroll
  for (int ch = 0; ch < 2; ++ch) {
    int ko = ch * 32;
    v16h a, bf;
#pragma unroll
    for (int v = 0; v < 8; ++v) {
      int kbA = 2 * v + 16 * half;
      a[2 * v]     = hrow[ko + kbA];
      a[2 * v + 1] = hrow[ko + kbA + 1];
      int kbB = 2 * v + 16 * half;
      bf[2 * v]     = (_Float16)wl[n * 64 + ko + kbB];
      bf[2 * v + 1] = (_Float16)wl[n * 64 + ko + kbB + 1];
    }
    c = WMMA_F16(a, bf, c);
  }
  float bias = bl[n];
#pragma unroll
  for (int v = 0; v < 8; ++v) {
    int row = rt * 16 + v + 8 * half;
    logits[(size_t)row * 64 + n] = c[v] + bias;
  }
}

// ---------------------------------------------------------------------------
// K4: sequential gumbel-softmax scan. One thread per batch row.
// Writes y_seq to workspace and log_q / log_p straight into d_out.
// ---------------------------------------------------------------------------
__global__ void yscan_kernel(const float* __restrict__ logits, const _Float16* __restrict__ hseq,
                             const float* __restrict__ wi, const float* __restrict__ bi,
                             const float* __restrict__ u0, const float* __restrict__ useq,
                             float* __restrict__ yseq,
                             float* __restrict__ outLq, float* __restrict__ outLp) {
  int b = threadIdx.x;
  const float TAUi = 2.0f;  // 1/TAU

  // init_logits = h_seq[0] @ wi^T + bi
  float il[8];
  const _Float16* h0 = hseq + (size_t)b * 64;
#pragma unroll
  for (int k = 0; k < 8; ++k) {
    float s = bi[k];
    for (int d = 0; d < 64; ++d) s = fmaf((float)h0[d], wi[k * 64 + d], s);
    il[k] = s;
  }

  float y[8];
  {
    float lg[8], mx = -1e30f;
    for (int k = 0; k < 8; ++k) {
      lg[k] = (il[k] + gumbelf_(u0[b * 8 + k])) * TAUi;
      mx = fmaxf(mx, lg[k]);
    }
    float se = 0.f;
    for (int k = 0; k < 8; ++k) { y[k] = __expf(lg[k] - mx); se += y[k]; }
    float inv = 1.f / se;
    for (int k = 0; k < 8; ++k) y[k] *= inv;

    float m2 = -1e30f;
    for (int k = 0; k < 8; ++k) m2 = fmaxf(m2, il[k]);
    float s2 = 0.f;
    for (int k = 0; k < 8; ++k) s2 += __expf(il[k] - m2);
    float lse = m2 + __logf(s2);
    const float logK = __logf(0.125f);
    float lq = 0.f, lp = 0.f;
    for (int k = 0; k < 8; ++k) { lq += y[k] * (il[k] - lse); lp += y[k] * logK; }
    for (int k = 0; k < 8; ++k) yseq[(size_t)b * 8 + k] = y[k];
    outLq[(size_t)b * Tn] = lq;
    outLp[(size_t)b * Tn] = lp;
  }

  const float cOff = (1.f - 0.9f) / 7.f;  // off-diagonal transition prob
  for (int t = 1; t < Tn; ++t) {
    const float* lgm = logits + ((size_t)t * Bsz + b) * 64;  // [8][8] row-major
    float lt[8];
#pragma unroll
    for (int j = 0; j < 8; ++j) {
      float s = 0.f;
#pragma unroll
      for (int k = 0; k < 8; ++k) s = fmaf(y[k], lgm[k * 8 + j], s);
      lt[j] = s;
    }
    float yn[8], lg2[8], mx = -1e30f;
    for (int j = 0; j < 8; ++j) {
      lg2[j] = (lt[j] + gumbelf_(useq[((size_t)b * (Tn - 1) + (t - 1)) * 8 + j])) * TAUi;
      mx = fmaxf(mx, lg2[j]);
    }
    float se = 0.f;
    for (int j = 0; j < 8; ++j) { yn[j] = __expf(lg2[j] - mx); se += yn[j]; }
    float inv = 1.f / se;
    for (int j = 0; j < 8; ++j) yn[j] *= inv;

    float m2 = -1e30f;
    for (int j = 0; j < 8; ++j) m2 = fmaxf(m2, lt[j]);
    float s2 = 0.f;
    for (int j = 0; j < 8; ++j) s2 += __expf(lt[j] - m2);
    float lse = m2 + __logf(s2);

    float lq = 0.f, lp = 0.f;
    for (int j = 0; j < 8; ++j) {
      lq += yn[j] * (lt[j] - lse);
      float tp = cOff + (0.9f - cOff) * y[j];  // y_prev @ trans (sum y == 1)
      lp += yn[j] * __logf(fmaxf(tp, 1e-8f));
    }
    for (int j = 0; j < 8; ++j) {
      y[j] = yn[j];
      yseq[((size_t)t * Bsz + b) * 8 + j] = yn[j];
    }
    outLq[(size_t)b * Tn + t] = lq;
    outLp[(size_t)b * Tn + t] = lp;
  }
}

// ---------------------------------------------------------------------------
// K5: output expansion (store-bandwidth bound, ~302 MB). One block per (b,t);
// 288 b128 NON-TEMPORAL stores (write-once stream; don't thrash 192MB L2):
// A(64) + Q(64) + B(32) + C-broadcast(128). Inputs stay RT-cached.
// ---------------------------------------------------------------------------
__global__ void expand_kernel(const float* __restrict__ yseq,
                              const v4f* __restrict__ A4, const v4f* __restrict__ B4,
                              const v4f* __restrict__ C4, const v4f* __restrict__ Q4,
                              v4f* __restrict__ oA, v4f* __restrict__ oB,
                              v4f* __restrict__ oC, v4f* __restrict__ oQ) {
  int idx = blockIdx.x;
  int b = idx >> 9, t = idx & 511;
  __shared__ float sy[8];
  if (threadIdx.x < 8) sy[threadIdx.x] = yseq[((size_t)t * Bsz + b) * 8 + threadIdx.x];
  __syncthreads();
  float yv[8];
#pragma unroll
  for (int k = 0; k < 8; ++k) yv[k] = sy[k];

  size_t bt = (size_t)b * Tn + t;
  for (int i = threadIdx.x; i < 288; i += 256) {
    if (i < 64) {
      int e = i;
      v4f acc = {};
#pragma unroll
      for (int k = 0; k < 8; ++k) acc += A4[k * 64 + e] * yv[k];
      __builtin_nontemporal_store(acc, &oA[bt * 64 + e]);
    } else if (i < 128) {
      int e = i - 64;
      v4f acc = {};
#pragma unroll
      for (int k = 0; k < 8; ++k) acc += Q4[k * 64 + e] * yv[k];
      __builtin_nontemporal_store(acc, &oQ[bt * 64 + e]);
    } else if (i < 160) {
      int e = i - 128;
      v4f acc = {};
#pragma unroll
      for (int k = 0; k < 8; ++k) acc += B4[k * 32 + e] * yv[k];
      __builtin_nontemporal_store(acc, &oB[bt * 32 + e]);
    } else {
      int e = i - 160;  // 0..127: broadcast C[0]
      v4f cv = C4[e];
      __builtin_nontemporal_store(cv, &oC[bt * 128 + e]);
    }
  }
}

// ---------------------------------------------------------------------------
extern "C" void kernel_launch(void* const* d_in, const int* in_sizes, int n_in,
                              void* d_out, int out_size, void* d_ws, size_t ws_size,
                              hipStream_t stream) {
  const float* a_seq = (const float*)d_in[0];
  const float* Amat  = (const float*)d_in[1];
  const float* Bmat  = (const float*)d_in[2];
  const float* Cmat  = (const float*)d_in[3];
  const float* Qmat  = (const float*)d_in[4];
  const float* wih_f = (const float*)d_in[5];
  const float* whh_f = (const float*)d_in[6];
  const float* bih_f = (const float*)d_in[7];
  const float* bhh_f = (const float*)d_in[8];
  const float* wih_b = (const float*)d_in[9];
  const float* whh_b = (const float*)d_in[10];
  const float* bih_b = (const float*)d_in[11];
  const float* bhh_b = (const float*)d_in[12];
  const float* wl    = (const float*)d_in[13];
  const float* bl    = (const float*)d_in[14];
  const float* wi    = (const float*)d_in[15];
  const float* bi    = (const float*)d_in[16];
  const float* u0    = (const float*)d_in[17];
  const float* useq  = (const float*)d_in[18];

  // Workspace layout (~74 MB total, every byte written before read):
  char* w = (char*)d_ws;
  float*    gxf    = (float*)(w);                    // 4096*6*256 f32 = 25,165,824 B (C-layout)
  float*    gxb    = (float*)(w + 25165824);         // 25,165,824 B
  _Float16* hseq   = (_Float16*)(w + 50331648);      // 512*128*64 f16 = 8,388,608 B
  float*    logits = (float*)(w + 58720256);         // 512*128*64 f32 = 16,777,216 B
  float*    yseq   = (float*)(w + 75497472);         // 512*128*8  f32 = 2,097,152 B

  // Output layout (flat concat, f32): A | B | C | Q | log_q | log_p
  float* out   = (float*)d_out;
  float* outA  = out;               // 128*512*256
  float* outB  = out + 16777216;    // 128*512*128
  float* outC  = out + 25165824;    // 128*512*512
  float* outQ  = out + 58720256;    // 128*512*256
  float* outLq = out + 75497472;    // 128*512
  float* outLp = out + 75563008;    // 128*512

  hipLaunchKernelGGL(gx_kernel, dim3(49152), dim3(32), 0, stream,
                     a_seq, wih_f, bih_f, wih_b, bih_b, gxf, gxb);
  hipLaunchKernelGGL(gru_kernel, dim3(16), dim3(32), 0, stream,
                     whh_f, bhh_f, whh_b, bhh_b, gxf, gxb, hseq);
  hipLaunchKernelGGL(logits_kernel, dim3(16384), dim3(32), 0, stream,
                     hseq, wl, bl, logits);
  hipLaunchKernelGGL(yscan_kernel, dim3(1), dim3(128), 0, stream,
                     logits, hseq, wi, bi, u0, useq, yseq, outLq, outLp);
  hipLaunchKernelGGL(expand_kernel, dim3(65536), dim3(256), 0, stream,
                     yseq, (const v4f*)Amat, (const v4f*)Bmat,
                     (const v4f*)Cmat, (const v4f*)Qmat,
                     (v4f*)outA, (v4f*)outB, (v4f*)outC, (v4f*)outQ);
}
